// GraphCast_20486994002522
// MI455X (gfx1250) — matrix operand order repaired and checked by reading, not compiled, not run on hardware
//
#include <hip/hip_runtime.h>
#include <hip/hip_bf16.h>

// ---------------------------------------------------------------------------
// GraphCast grid->mesh block on MI455X (gfx1250, wave32, WMMA).
// Three fused MLP kernels (edge / mesh-node / grid-node), each:
//   gather+concat -> bf16 LDS -> WMMA GEMM1 -> bias+SiLU -> WMMA GEMM2
//   -> bias -> LayerNorm -> (scatter-atomic | residual store)
// 8 waves / block; each 16-row strip is computed by a pair of waves that
// split the 8 N-tiles (4 each) to keep accumulator pressure at 32 VGPRs.
// Weights are staged into LDS pre-swizzled into the WMMA B-fragment layout
// so each fragment load is 2x ds_load_b128 per lane.
// ---------------------------------------------------------------------------

static constexpr int HDIM  = 128;   // hidden width
static constexpr int TROWS = 64;    // rows per block
static constexpr int TPB   = 256;   // 8 waves of 32

typedef __attribute__((ext_vector_type(16))) __bf16 v16bf;
typedef __attribute__((ext_vector_type(8)))  __bf16 v8bf;
typedef __attribute__((ext_vector_type(8)))  float  v8f;

static __device__ __forceinline__ v8f wmma_bf16(v16bf a, v16bf b, v8f c) {
    // v_wmma_f32_16x16x32_bf16 (neg_a, A, neg_b, B, c_mod, C, reuse_a, reuse_b)
    return __builtin_amdgcn_wmma_f32_16x16x32_bf16(false, a, false, b, (short)0, c, false, false);
}

static __device__ __forceinline__ unsigned pack_bf2(float x, float y) {
    unsigned short lo = __builtin_bit_cast(unsigned short, (__bf16)x);
    unsigned short hi = __builtin_bit_cast(unsigned short, (__bf16)y);
    return (unsigned)lo | ((unsigned)hi << 16);
}

// A fragment: 16x32 bf16, rows [rowBase, rowBase+16) of an LDS matrix (stride lda).
// Contiguous K order (matches swizzled B): lanes 0-15 -> M=lane, halves 0..15 = K(k0..k0+15);
//             lanes 16-31 -> M=lane-16, halves 0..15 = K(k0+16..k0+31).
// lda is a multiple of 8 and k0 a multiple of 32 -> both loads are 16B aligned.
static __device__ __forceinline__ v16bf load_a_frag(const __bf16* base, int lda,
                                                    int rowBase, int k0) {
    const int lane = threadIdx.x & 31;
    const int m    = lane & 15;
    const int kb   = (lane & 16) ? 16 : 0;
    const __bf16* p = base + (rowBase + m) * lda + k0 + kb;
    v8bf lo = *(const v8bf*)p;          // ds_load_b128
    v8bf hi = *(const v8bf*)(p + 8);    // ds_load_b128
    return __builtin_shufflevector(lo, hi, 0, 1, 2, 3, 4, 5, 6, 7,
                                           8, 9, 10, 11, 12, 13, 14, 15);
}

// Stage a 128xHDIM fp32 weight chunk into LDS, pre-swizzled to B-fragment order:
//   ldsW[ ((kt*8+nt)*32 + lane)*16 + j ] = (bf16) W[kt*32 + (lane>>4)*16 + j][nt*16 + (lane&15)]
// Each thread converts pairs along K and emits one b32 LDS store.
static __device__ __forceinline__ void stage_w_swz(__bf16* ldsW,
                                                   const float* __restrict__ W) {
    unsigned* ldsW32 = (unsigned*)ldsW;
    for (int idx = threadIdx.x; idx < (HDIM / 2) * HDIM; idx += TPB) {
        const int k  = (idx >> 7) << 1;       // even k
        const int n  = idx & (HDIM - 1);
        const float w0 = W[k * HDIM + n];
        const float w1 = W[(k + 1) * HDIM + n];
        const int kt = k >> 5, kb = (k >> 4) & 1, j = k & 15;
        const int nt = n >> 4, nl = n & 15;
        const int lane = (kb << 4) | nl;
        const int ha = ((((kt << 3) | nt) << 5 | lane) << 4) | j;  // half index (even)
        ldsW32[ha >> 1] = pack_bf2(w0, w1);
    }
}

// B fragment: one 32-byte contiguous per-lane read from the swizzled tile.
static __device__ __forceinline__ v16bf load_b_frag_swz(const __bf16* ldsW,
                                                        int kt, int nt) {
    const int lane = threadIdx.x & 31;
    return *(const v16bf*)(ldsW + ((((kt << 3) | nt) << 5 | lane) << 4));
}

// Shared 2-layer MLP core: ldsCat[TROWS x IN_DIM] (bf16, stride IN_DIM+8)
//  -> GEMM1 (+b1, SiLU) -> ldsH -> GEMM2 (+b2) -> ldsY (f32, stride HDIM+4).
// Wave w handles rows [ (w>>1)*16, +16 ) and N-tiles [ (w&1)*4, +4 ).
template <int IN_DIM>
static __device__ void mlp_core(const __bf16* ldsCat, __bf16* ldsW, __bf16* ldsH,
                                float* ldsY,
                                const float* __restrict__ W1, const float* __restrict__ b1,
                                const float* __restrict__ W2, const float* __restrict__ b2) {
    const int tid     = threadIdx.x;
    const int lane    = tid & 31;
    const int nlow    = lane & 15;
    const int mo      = (lane & 16) ? 8 : 0;
    const int wave    = tid >> 5;
    const int rowBase = (wave >> 1) << 4;   // 16-row strip (2 waves share it)
    const int ntBase  = (wave & 1) << 2;    // N-tiles [ntBase, ntBase+4)
    constexpr int LDA = IN_DIM + 8;
    constexpr int LDH = HDIM + 8;
    constexpr int LDY = HDIM + 4;
    constexpr int KCHUNKS = IN_DIM / HDIM;

    v8f acc[4];
#pragma unroll
    for (int i = 0; i < 4; ++i)
        acc[i] = v8f{0.f, 0.f, 0.f, 0.f, 0.f, 0.f, 0.f, 0.f};

    // ---- GEMM1: [TROWS, IN_DIM] x [IN_DIM, HDIM], K staged in 128-chunks ----
    for (int kk = 0; kk < KCHUNKS; ++kk) {
        stage_w_swz(ldsW, W1 + (size_t)kk * HDIM * HDIM);
        __syncthreads();
#pragma unroll
        for (int kt = 0; kt < 4; ++kt) {
            v16bf a = load_a_frag(ldsCat, LDA, rowBase, kk * HDIM + kt * 32);
#pragma unroll
            for (int nt = 0; nt < 4; ++nt)
                acc[nt] = wmma_bf16(a, load_b_frag_swz(ldsW, kt, ntBase + nt), acc[nt]);
        }
        __syncthreads();
    }

    // ---- bias + SiLU -> ldsH (bf16), C-layout scatter ----
#pragma unroll
    for (int nt = 0; nt < 4; ++nt) {
        const int ncol = (ntBase + nt) * 16 + nlow;
        const float bias = b1[ncol];
#pragma unroll
        for (int r = 0; r < 8; ++r) {
            float x = acc[nt][r] + bias;
            float s = x / (1.f + __expf(-x));
            ldsH[(rowBase + r + mo) * LDH + ncol] = (__bf16)s;
        }
    }

    // ---- stage W2 (swizzled), GEMM2: [TROWS, HDIM] x [HDIM, HDIM] ----
    stage_w_swz(ldsW, W2);
    __syncthreads();

    v8f acc2[4];
#pragma unroll
    for (int i = 0; i < 4; ++i)
        acc2[i] = v8f{0.f, 0.f, 0.f, 0.f, 0.f, 0.f, 0.f, 0.f};
#pragma unroll
    for (int kt = 0; kt < 4; ++kt) {
        v16bf a = load_a_frag(ldsH, LDH, rowBase, kt * 32);
#pragma unroll
        for (int nt = 0; nt < 4; ++nt)
            acc2[nt] = wmma_bf16(a, load_b_frag_swz(ldsW, kt, ntBase + nt), acc2[nt]);
    }
#pragma unroll
    for (int nt = 0; nt < 4; ++nt) {
        const int ncol = (ntBase + nt) * 16 + nlow;
        const float bias = b2[ncol];
#pragma unroll
        for (int r = 0; r < 8; ++r)
            ldsY[(rowBase + r + mo) * LDY + ncol] = acc2[nt][r] + bias;
    }
    __syncthreads();
}

// Per-row LayerNorm statistics over HDIM columns of ldsY (float4 scans).
static __device__ __forceinline__ void ln_stats(const float* ldsY, float* ldsMu, float* ldsRs) {
    constexpr int LDY = HDIM + 4;
    const int tid = threadIdx.x;
    if (tid < TROWS) {
        float s = 0.f, s2 = 0.f;
        for (int c = 0; c < HDIM; c += 4) {
            float4 v = *(const float4*)(ldsY + tid * LDY + c);
            s  += v.x + v.y + v.z + v.w;
            s2 += v.x * v.x + v.y * v.y + v.z * v.z + v.w * v.w;
        }
        float mu  = s * (1.f / HDIM);
        float var = s2 * (1.f / HDIM) - mu * mu;
        ldsMu[tid] = mu;
        ldsRs[tid] = rsqrtf(var + 1e-5f);
    }
    __syncthreads();
}

// ---------------------------------------------------------------------------
// Kernel 1: edge MLP (in = 3H) + LayerNorm + segment-sum scatter into agg.
// ---------------------------------------------------------------------------
__global__ void __launch_bounds__(TPB)
edge_mlp_scatter(const float* __restrict__ efeat, const float* __restrict__ grid,
                 const float* __restrict__ mesh, const int* __restrict__ src,
                 const int* __restrict__ dst,
                 const float* __restrict__ W1, const float* __restrict__ b1,
                 const float* __restrict__ W2, const float* __restrict__ b2,
                 const float* __restrict__ g,  const float* __restrict__ bln,
                 float* __restrict__ agg, long long E) {
    constexpr int IN_DIM = 3 * HDIM;
    constexpr int LDA = IN_DIM + 8;
    constexpr int LDY = HDIM + 4;
    constexpr int QPR = IN_DIM / 4;     // float4 quads per row
    extern __shared__ char smem[];
    __bf16* ldsCat = (__bf16*)smem;
    __bf16* ldsW   = ldsCat + TROWS * LDA;
    __bf16* ldsH   = ldsW + HDIM * HDIM;
    float*  ldsY   = (float*)(ldsH + TROWS * (HDIM + 8));
    float*  ldsMu  = ldsY + TROWS * LDY;
    float*  ldsRs  = ldsMu + TROWS;

    const int tid = threadIdx.x;
    const long long e0 = (long long)blockIdx.x * TROWS;

    // gather + concat [efeat | grid[src] | mesh[dst]] -> bf16 LDS (float4 reads)
    for (int q = tid; q < TROWS * QPR; q += TPB) {
        const int r = q / QPR;
        const int c = (q - r * QPR) * 4;
        long long e = e0 + r; if (e >= E) e = E - 1;
        const float* sp;
        if (c < HDIM)          sp = efeat + e * HDIM + c;
        else if (c < 2 * HDIM) sp = grid + (long long)src[e] * HDIM + (c - HDIM);
        else                   sp = mesh + (long long)dst[e] * HDIM + (c - 2 * HDIM);
        float4 v = *(const float4*)sp;
        unsigned* dp = (unsigned*)(ldsCat + r * LDA + c);
        dp[0] = pack_bf2(v.x, v.y);
        dp[1] = pack_bf2(v.z, v.w);
    }
    __syncthreads();

    mlp_core<IN_DIM>(ldsCat, ldsW, ldsH, ldsY, W1, b1, W2, b2);
    ln_stats(ldsY, ldsMu, ldsRs);

    // normalized edge feature -> atomic segment-sum into agg[dst]
    for (int idx = tid; idx < TROWS * HDIM; idx += TPB) {
        const int r = idx >> 7, c = idx & (HDIM - 1);
        const long long e = e0 + r;
        if (e < E) {
            float v = (ldsY[r * LDY + c] - ldsMu[r]) * ldsRs[r] * g[c] + bln[c];
            __hip_atomic_fetch_add(&agg[(long long)dst[e] * HDIM + c], v,
                                   __ATOMIC_RELAXED, __HIP_MEMORY_SCOPE_AGENT);
        }
    }
}

// ---------------------------------------------------------------------------
// Kernel 2: mesh-node MLP (in = [agg | mesh], 2H) + LN + residual.
// ---------------------------------------------------------------------------
__global__ void __launch_bounds__(TPB)
node_mlp_mesh(const float* __restrict__ agg, const float* __restrict__ mesh,
              const float* __restrict__ W1, const float* __restrict__ b1,
              const float* __restrict__ W2, const float* __restrict__ b2,
              const float* __restrict__ g,  const float* __restrict__ bln,
              float* __restrict__ out, long long NM) {
    constexpr int IN_DIM = 2 * HDIM;
    constexpr int LDA = IN_DIM + 8;
    constexpr int LDY = HDIM + 4;
    constexpr int QPR = IN_DIM / 4;
    extern __shared__ char smem[];
    __bf16* ldsCat = (__bf16*)smem;
    __bf16* ldsW   = ldsCat + TROWS * LDA;
    __bf16* ldsH   = ldsW + HDIM * HDIM;
    float*  ldsY   = (float*)(ldsH + TROWS * (HDIM + 8));
    float*  ldsMu  = ldsY + TROWS * LDY;
    float*  ldsRs  = ldsMu + TROWS;

    const int tid = threadIdx.x;
    const long long r0 = (long long)blockIdx.x * TROWS;

    for (int q = tid; q < TROWS * QPR; q += TPB) {
        const int r = q >> 6;                 // QPR == 64
        const int c = (q & 63) * 4;
        long long row = r0 + r; if (row >= NM) row = NM - 1;
        const float* sp = (c < HDIM) ? (agg + row * HDIM + c)
                                     : (mesh + row * HDIM + (c - HDIM));
        float4 v = *(const float4*)sp;
        unsigned* dp = (unsigned*)(ldsCat + r * LDA + c);
        dp[0] = pack_bf2(v.x, v.y);
        dp[1] = pack_bf2(v.z, v.w);
    }
    __syncthreads();

    mlp_core<IN_DIM>(ldsCat, ldsW, ldsH, ldsY, W1, b1, W2, b2);
    ln_stats(ldsY, ldsMu, ldsRs);

    for (int q = tid; q < TROWS * (HDIM / 4); q += TPB) {
        const int r = q >> 5, c = (q & 31) * 4;
        const long long row = r0 + r;
        if (row < NM) {
            float4 y4 = *(const float4*)(ldsY + r * LDY + c);
            float4 g4 = *(const float4*)(g + c);
            float4 b4 = *(const float4*)(bln + c);
            float4 m4 = *(const float4*)(mesh + row * HDIM + c);
            const float mu = ldsMu[r], rs = ldsRs[r];
            float4 o;
            o.x = m4.x + (y4.x - mu) * rs * g4.x + b4.x;
            o.y = m4.y + (y4.y - mu) * rs * g4.y + b4.y;
            o.z = m4.z + (y4.z - mu) * rs * g4.z + b4.z;
            o.w = m4.w + (y4.w - mu) * rs * g4.w + b4.w;
            *(float4*)(out + row * HDIM + c) = o;
        }
    }
}

// ---------------------------------------------------------------------------
// Kernel 3: grid-node MLP (in = H) + LN + residual.
// ---------------------------------------------------------------------------
__global__ void __launch_bounds__(TPB)
node_mlp_grid(const float* __restrict__ grid,
              const float* __restrict__ W1, const float* __restrict__ b1,
              const float* __restrict__ W2, const float* __restrict__ b2,
              const float* __restrict__ g,  const float* __restrict__ bln,
              float* __restrict__ out, long long NG) {
    constexpr int IN_DIM = HDIM;
    constexpr int LDA = IN_DIM + 8;
    constexpr int LDY = HDIM + 4;
    extern __shared__ char smem[];
    __bf16* ldsCat = (__bf16*)smem;
    __bf16* ldsW   = ldsCat + TROWS * LDA;
    __bf16* ldsH   = ldsW + HDIM * HDIM;
    float*  ldsY   = (float*)(ldsH + TROWS * (HDIM + 8));
    float*  ldsMu  = ldsY + TROWS * LDY;
    float*  ldsRs  = ldsMu + TROWS;

    const int tid = threadIdx.x;
    const long long r0 = (long long)blockIdx.x * TROWS;

    for (int q = tid; q < TROWS * (HDIM / 4); q += TPB) {
        const int r = q >> 5, c = (q & 31) * 4;
        long long row = r0 + r; if (row >= NG) row = NG - 1;
        float4 v = *(const float4*)(grid + row * HDIM + c);
        unsigned* dp = (unsigned*)(ldsCat + r * LDA + c);
        dp[0] = pack_bf2(v.x, v.y);
        dp[1] = pack_bf2(v.z, v.w);
    }
    __syncthreads();

    mlp_core<IN_DIM>(ldsCat, ldsW, ldsH, ldsY, W1, b1, W2, b2);
    ln_stats(ldsY, ldsMu, ldsRs);

    for (int q = tid; q < TROWS * (HDIM / 4); q += TPB) {
        const int r = q >> 5, c = (q & 31) * 4;
        const long long row = r0 + r;
        if (row < NG) {
            float4 y4 = *(const float4*)(ldsY + r * LDY + c);
            float4 g4 = *(const float4*)(g + c);
            float4 b4 = *(const float4*)(bln + c);
            float4 x4 = *(const float4*)(grid + row * HDIM + c);
            const float mu = ldsMu[r], rs = ldsRs[r];
            float4 o;
            o.x = x4.x + (y4.x - mu) * rs * g4.x + b4.x;
            o.y = x4.y + (y4.y - mu) * rs * g4.y + b4.y;
            o.z = x4.z + (y4.z - mu) * rs * g4.z + b4.z;
            o.w = x4.w + (y4.w - mu) * rs * g4.w + b4.w;
            *(float4*)(out + row * HDIM + c) = o;
        }
    }
}

// ---------------------------------------------------------------------------
extern "C" void kernel_launch(void* const* d_in, const int* in_sizes, int n_in,
                              void* d_out, int out_size, void* d_ws, size_t ws_size,
                              hipStream_t stream) {
    const float* g2m  = (const float*)d_in[0];
    const float* grid = (const float*)d_in[1];
    const float* mesh = (const float*)d_in[2];
    const int*   src  = (const int*)d_in[3];
    const int*   dst  = (const int*)d_in[4];
    // d_in[5] = num_dst_nodes (derived from in_sizes instead)
    const float* eW1 = (const float*)d_in[6];  const float* eb1 = (const float*)d_in[7];
    const float* eW2 = (const float*)d_in[8];  const float* eb2 = (const float*)d_in[9];
    const float* eg  = (const float*)d_in[10]; const float* ebn = (const float*)d_in[11];
    const float* sW1 = (const float*)d_in[12]; const float* sb1 = (const float*)d_in[13];
    const float* sW2 = (const float*)d_in[14]; const float* sb2 = (const float*)d_in[15];
    const float* sg  = (const float*)d_in[16]; const float* sbn = (const float*)d_in[17];
    const float* dW1 = (const float*)d_in[18]; const float* db1 = (const float*)d_in[19];
    const float* dW2 = (const float*)d_in[20]; const float* db2 = (const float*)d_in[21];
    const float* dg  = (const float*)d_in[22]; const float* dbn = (const float*)d_in[23];

    const long long E  = (long long)in_sizes[0] / HDIM;   // 524288
    const long long NG = (long long)in_sizes[1] / HDIM;   // 262144
    const long long NM = (long long)in_sizes[2] / HDIM;   // 40962

    float* agg      = (float*)d_ws;            // [NM, H] segment-sum accumulator
    float* out_mesh = (float*)d_out;           // mesh_new: [NM, H]
    float* out_grid = out_mesh + NM * HDIM;    // grid_new: [NG, H]

    hipMemsetAsync(agg, 0, (size_t)NM * HDIM * sizeof(float), stream);

    auto sh = [](int in_dim) {
        return (size_t)TROWS * (in_dim + 8) * sizeof(__bf16)   // cat
             + (size_t)HDIM * HDIM * sizeof(__bf16)            // W tile (swizzled)
             + (size_t)TROWS * (HDIM + 8) * sizeof(__bf16)     // h
             + (size_t)TROWS * (HDIM + 4) * sizeof(float)      // y
             + (size_t)2 * TROWS * sizeof(float);              // mu, rs
    };

    edge_mlp_scatter<<<(int)((E + TROWS - 1) / TROWS), TPB, sh(3 * HDIM), stream>>>(
        g2m, grid, mesh, src, dst, eW1, eb1, eW2, eb2, eg, ebn, agg, E);

    node_mlp_grid<<<(int)((NG + TROWS - 1) / TROWS), TPB, sh(HDIM), stream>>>(
        grid, sW1, sb1, sW2, sb2, sg, sbn, out_grid, NG);

    node_mlp_mesh<<<(int)((NM + TROWS - 1) / TROWS), TPB, sh(2 * HDIM), stream>>>(
        agg, mesh, dW1, db1, dW2, db2, dg, dbn, out_mesh, NM);
}
